// GATLayer_37280316129311
// MI455X (gfx1250) — matrix operand order repaired and checked by reading, not compiled, not run on hardware
//
#include <hip/hip_runtime.h>
#include <hip/hip_bf16.h>

typedef __attribute__((ext_vector_type(16))) __bf16 v16bf;
typedef __attribute__((ext_vector_type(8)))  float  v8f;

#define IN_DIM  256
#define OUT_DIM 64
#define NEG_SLOPE 0.01f

// ---------------------------------------------------------------------------
// Pack W [256][64] f32 -> bf16 in WMMA B-fragment order so each lane's 16
// values for a (nt, kstep) fragment are contiguous (one 32B load).
// index = ((nt*8 + kstep)*32 + lane)*16 + j
//   k = kstep*32 + ((j<8)?0:16) + ((lane&16)?8:0) + (j&7) ,  n = nt*16 + (lane&15)
// ---------------------------------------------------------------------------
__global__ void __launch_bounds__(256)
pack_w_bf16(const float* __restrict__ W, __bf16* __restrict__ Wfrag) {
  int t = blockIdx.x * blockDim.x + threadIdx.x;
  if (t >= IN_DIM * OUT_DIM) return;
  int j     = t & 15;
  int lane  = (t >> 4) & 31;
  int kstep = (t >> 9) & 7;
  int nt    = t >> 12;
  int k = kstep * 32 + ((j < 8) ? 0 : 16) + ((lane & 16) ? 8 : 0) + (j & 7);
  int n = nt * 16 + (lane & 15);
  Wfrag[t] = (__bf16)W[k * OUT_DIM + n];
}

// ---------------------------------------------------------------------------
// z = h @ W via v_wmma_f32_16x16x32_bf16. One wave handles a 16-row tile and
// all 64 output columns (4 accumulators). EXEC stays all-ones (row clamp on
// load, predicate on store) as WMMA requires.
// ---------------------------------------------------------------------------
__global__ void __launch_bounds__(256)
gat_gemm_wmma(const float* __restrict__ h, const __bf16* __restrict__ Wfrag,
              float* __restrict__ z, int n_nodes) {
  const int lane   = threadIdx.x & 31;
  const int mtile  = blockIdx.x * 8 + (threadIdx.x >> 5);
  const int ntiles = (n_nodes + 15) >> 4;
  if (mtile >= ntiles) return;          // wave-uniform exit

  int row = mtile * 16 + (lane & 15);   // A row this lane supplies
  if (row >= n_nodes) row = n_nodes - 1;
  const float* hrow = h + (size_t)row * IN_DIM;
  const int kb  = (lane & 16) ? 8 : 0;  // A layout: lanes0-15 K[0..7], lanes16-31 K[8..15]
  const int kb2 = kb + 16;              //           then K[16..23] / K[24..31]

  v8f acc0 = {}, acc1 = {}, acc2 = {}, acc3 = {};

  for (int kstep = 0; kstep < 8; ++kstep) {
    const int K0 = kstep << 5;
    const float4* pa0 = (const float4*)(hrow + K0 + kb);
    const float4* pa1 = (const float4*)(hrow + K0 + kb2);
    float4 f0 = pa0[0], f1 = pa0[1];
    float4 f2 = pa1[0], f3 = pa1[1];
    v16bf a;
    a[0] = (__bf16)f0.x; a[1] = (__bf16)f0.y; a[2] = (__bf16)f0.z; a[3] = (__bf16)f0.w;
    a[4] = (__bf16)f1.x; a[5] = (__bf16)f1.y; a[6] = (__bf16)f1.z; a[7] = (__bf16)f1.w;
    a[8] = (__bf16)f2.x; a[9] = (__bf16)f2.y; a[10] = (__bf16)f2.z; a[11] = (__bf16)f2.w;
    a[12] = (__bf16)f3.x; a[13] = (__bf16)f3.y; a[14] = (__bf16)f3.z; a[15] = (__bf16)f3.w;

    const __bf16* base = Wfrag + ((size_t)(kstep * 32 + lane) << 4);
    v16bf b;
    b = *(const v16bf*)(base);
    acc0 = __builtin_amdgcn_wmma_f32_16x16x32_bf16(false, a, false, b, (short)0, acc0, false, false);
    b = *(const v16bf*)(base + 4096);   // nt stride = 8*32*16 elements
    acc1 = __builtin_amdgcn_wmma_f32_16x16x32_bf16(false, a, false, b, (short)0, acc1, false, false);
    b = *(const v16bf*)(base + 8192);
    acc2 = __builtin_amdgcn_wmma_f32_16x16x32_bf16(false, a, false, b, (short)0, acc2, false, false);
    b = *(const v16bf*)(base + 12288);
    acc3 = __builtin_amdgcn_wmma_f32_16x16x32_bf16(false, a, false, b, (short)0, acc3, false, false);
  }

  // C/D layout: n = lane&15, m = j + ((lane&16)?8:0)
  const int n0    = lane & 15;
  const int mbase = mtile * 16 + ((lane & 16) ? 8 : 0);
  for (int j = 0; j < 8; ++j) {
    int r = mbase + j;
    if (r < n_nodes) {
      float* zr = z + (size_t)r * OUT_DIM + n0;
      zr[0]  = acc0[j];
      zr[16] = acc1[j];
      zr[32] = acc2[j];
      zr[48] = acc3[j];
    }
  }
}

// ---------------------------------------------------------------------------
// s_src[i] = z[i] . a[:64] ; s_dst[i] = z[i] . a[64:]
// ---------------------------------------------------------------------------
__global__ void __launch_bounds__(256)
attn_scores(const float* __restrict__ z, const float* __restrict__ a_vec,
            float* __restrict__ s_src, float* __restrict__ s_dst, int n) {
  int i = blockIdx.x * blockDim.x + threadIdx.x;
  if (i >= n) return;
  const float4* zr  = (const float4*)(z + (size_t)i * OUT_DIM);
  const float4* alo = (const float4*)(a_vec);
  const float4* ahi = (const float4*)(a_vec + OUT_DIM);
  float ss = 0.f, sd = 0.f;
  for (int j = 0; j < OUT_DIM / 4; ++j) {
    float4 zv = zr[j], av = alo[j], bv = ahi[j];
    ss += zv.x * av.x + zv.y * av.y + zv.z * av.z + zv.w * av.w;
    sd += zv.x * bv.x + zv.y * bv.y + zv.z * bv.z + zv.w * bv.w;
  }
  s_src[i] = ss;
  s_dst[i] = sd;
}

// ---------------------------------------------------------------------------
// Zero out, e_max (ordered-uint encoding: 0 == "below -inf"), denom.
// ---------------------------------------------------------------------------
__global__ void __launch_bounds__(256)
init_buffers(unsigned* __restrict__ e_max, float* __restrict__ denom,
             float* __restrict__ out, int n) {
  int i = blockIdx.x * blockDim.x + threadIdx.x;
  if (i < n * OUT_DIM) out[i] = 0.f;
  if (i < n) { e_max[i] = 0u; denom[i] = 0.f; }
}

// monotone float<->uint order encoding for atomicMax on floats
__device__ __forceinline__ unsigned ford_enc(float x) {
  unsigned u = __float_as_uint(x);
  return u ^ ((unsigned)((int)u >> 31) | 0x80000000u);
}
__device__ __forceinline__ float ford_dec(unsigned e) {
  unsigned mask = (e & 0x80000000u) ? 0x80000000u : 0xFFFFFFFFu;
  return __uint_as_float(e ^ mask);
}

__global__ void __launch_bounds__(256)
edge_logits(const float* __restrict__ s_src, const float* __restrict__ s_dst,
            const int* __restrict__ src, const int* __restrict__ dst,
            float* __restrict__ e_buf, unsigned* __restrict__ e_max, int E) {
  int i = blockIdx.x * blockDim.x + threadIdx.x;
  if (i >= E) return;
  float e = s_src[src[i]] + s_dst[dst[i]];
  e = (e > 0.f) ? e : NEG_SLOPE * e;           // leaky_relu
  e_buf[i] = e;
  atomicMax(&e_max[dst[i]], ford_enc(e));       // segment max
}

__global__ void __launch_bounds__(256)
edge_exp(const int* __restrict__ dst, float* __restrict__ e_buf,
         const unsigned* __restrict__ e_max, float* __restrict__ denom, int E) {
  int i = blockIdx.x * blockDim.x + threadIdx.x;
  if (i >= E) return;
  int d = dst[i];
  float p = __expf(e_buf[i] - ford_dec(e_max[d]));
  e_buf[i] = p;                                 // in-place: exp values
  __hip_atomic_fetch_add(&denom[d], p, __ATOMIC_RELAXED, __HIP_MEMORY_SCOPE_AGENT);
}

// 64 lanes per edge: out[dst] += alpha * z[src], coalesced row-wise atomics.
__global__ void __launch_bounds__(256)
edge_scatter(const int* __restrict__ src, const int* __restrict__ dst,
             const float* __restrict__ e_buf, const float* __restrict__ denom,
             const float* __restrict__ z, float* __restrict__ out, int total) {
  int t = blockIdx.x * blockDim.x + threadIdx.x;
  if (t >= total) return;
  int i = t >> 6;
  int f = t & 63;
  int d = dst[i], s = src[i];
  float alpha = e_buf[i] / denom[d];
  __hip_atomic_fetch_add(&out[(size_t)d * OUT_DIM + f],
                         alpha * z[(size_t)s * OUT_DIM + f],
                         __ATOMIC_RELAXED, __HIP_MEMORY_SCOPE_AGENT);
}

// ---------------------------------------------------------------------------
extern "C" void kernel_launch(void* const* d_in, const int* in_sizes, int n_in,
                              void* d_out, int out_size, void* d_ws, size_t ws_size,
                              hipStream_t stream) {
  const float* h     = (const float*)d_in[0];
  const int*   src   = (const int*)d_in[1];
  const int*   dst   = (const int*)d_in[2];
  const float* W     = (const float*)d_in[3];
  const float* a_vec = (const float*)d_in[4];
  float* out = (float*)d_out;

  const int n = in_sizes[0] / IN_DIM;   // 50000
  const int E = in_sizes[1];            // 800000

  // workspace carve-up (Wfrag first -> 32B alignment for v16bf loads)
  char* ws = (char*)d_ws;
  __bf16*   Wfrag = (__bf16*)ws;   ws += (size_t)IN_DIM * OUT_DIM * sizeof(__bf16);
  float*    z     = (float*)ws;    ws += (size_t)n * OUT_DIM * sizeof(float);
  float*    s_src = (float*)ws;    ws += (size_t)n * sizeof(float);
  float*    s_dst = (float*)ws;    ws += (size_t)n * sizeof(float);
  unsigned* e_max = (unsigned*)ws; ws += (size_t)n * sizeof(unsigned);
  float*    denom = (float*)ws;    ws += (size_t)n * sizeof(float);
  float*    e_buf = (float*)ws;    // E floats

  pack_w_bf16<<<(IN_DIM * OUT_DIM + 255) / 256, 256, 0, stream>>>(W, Wfrag);

  const int ntiles = (n + 15) / 16;
  gat_gemm_wmma<<<(ntiles + 7) / 8, 256, 0, stream>>>(h, Wfrag, z, n);

  attn_scores<<<(n + 255) / 256, 256, 0, stream>>>(z, a_vec, s_src, s_dst, n);

  init_buffers<<<((size_t)n * OUT_DIM + 255) / 256, 256, 0, stream>>>(e_max, denom, out, n);

  edge_logits<<<(E + 255) / 256, 256, 0, stream>>>(s_src, s_dst, src, dst, e_buf, e_max, E);
  edge_exp<<<(E + 255) / 256, 256, 0, stream>>>(dst, e_buf, e_max, denom, E);

  const int total = E * OUT_DIM;        // 51.2M, fits int32
  edge_scatter<<<(total + 255) / 256, 256, 0, stream>>>(src, dst, e_buf, denom, z, out, total);
}
